// NeuralODEDecoder_25202868093490
// MI455X (gfx1250) — matrix operand order repaired and checked by reading, not compiled, not run on hardware
//
#include <hip/hip_runtime.h>
#include <hip/hip_bf16.h>

// ---------------------------------------------------------------------------
// Neural ODE decoder for MI455X (gfx1250), wave32 + v_wmma_f32_16x16x32_bf16.
// Persistent kernel: 16 batch rows per WG, z state resident in C-fragments.
// ---------------------------------------------------------------------------

#define LATENT 256
#define HIDDEN 1024
#define OUTF   512
#define NSTEPS 10
#define BM     16          // batch rows per workgroup
#define NWAVE  8
#define BLOCK  (NWAVE * 32)

typedef __bf16 bf16_t;
typedef bf16_t v16bf __attribute__((ext_vector_type(16)));
typedef bf16_t v8bf  __attribute__((ext_vector_type(8)));
typedef float  v8f   __attribute__((ext_vector_type(8)));

// LDS strides (in elements), padded so row offsets are 16B aligned and
// A-fragment b128 loads hit distinct bank groups per row.
#define ZB_STRIDE 264      // bf16, z tile 16x256     (528B/row -> 4-bank shift)
#define H1_STRIDE 1048     // bf16, h tile 16x1024    (2096B/row -> 12-bank shift)
#define DZ_STRIDE 260      // f32,  dz tile 16x256
#define SC_STRIDE 40       // bf16, per-wave 16x32 scratch (80B/row)

// LDS layout (bytes). Region0 time-shares z_bf16 (8448B) with per-wave h2
// scratch (10240B): z_bf16 is only read in layer1, scratch only used in the
// fused layer2/3 phase after a barrier, and z_bf16 is rewritten at step end
// after another barrier.
#define LDS_REGION0 (NWAVE * 16 * SC_STRIDE * 2)       // 10240
#define LDS_H1_OFF  LDS_REGION0
#define LDS_DZ_OFF  (LDS_H1_OFF + 16 * H1_STRIDE * 2)  // +33536
#define LDS_TOTAL   (LDS_DZ_OFF + 16 * DZ_STRIDE * 4)  // +16640 = 60416

static __device__ __forceinline__ bf16_t f2bf(float f) {
    unsigned u = __float_as_uint(f);
    unsigned r = (u + 0x7FFFu + ((u >> 16) & 1u)) >> 16;   // round-nearest-even
    unsigned short s = (unsigned short)r;
    bf16_t b; __builtin_memcpy(&b, &s, 2); return b;
}

// A fragment (16x32 bf16) from a row-major LDS tile. Contiguous k order (same as pack_w_kernel):
// lane<16 holds row=lane, K = kt*32+0..15; lanes 16-31 hold
// the same rows with K offset by 16.
static __device__ __forceinline__ v16bf load_a_lds(const bf16_t* base, int stride,
                                                   int kt, int lane) {
    int row = lane & 15, kh = (lane >> 4) & 1;
    const bf16_t* q = base + row * stride + kt * 32 + kh * 16;
    v8bf lo = *(const v8bf*)q;
    v8bf hi = *(const v8bf*)(q + 8);
    return __builtin_shufflevector(lo, hi, 0,1,2,3,4,5,6,7,8,9,10,11,12,13,14,15);
}

// B fragment from pre-packed weights: tile (nt,kt) stored as 32 lanes x 16 bf16.
static __device__ __forceinline__ v16bf load_b_pk(const v16bf* pk, int ktiles,
                                                  int nt, int kt, int lane) {
    return pk[(size_t)(nt * ktiles + kt) * 32 + lane];
}

static __device__ __forceinline__ v8f wmma_bf16(v16bf a, v16bf b, v8f c) {
    return __builtin_amdgcn_wmma_f32_16x16x32_bf16(false, a, false, b,
                                                   (short)0, c, false, false);
}

static __device__ __forceinline__ float elu(float x) {
    return x > 0.0f ? x : (__expf(x) - 1.0f);
}

// ---------------------------------------------------------------------------
// Weight packing: W is row-major [N][K] (out = act @ W^T). B-matrix tile
// (nt,kt): lane l -> column n = nt*16 + (l&15), elements e -> k = kt*32 +
// (l>>4)*16 + e. 16 contiguous bf16 per (tile,lane) => coalesced b128 pairs.
// ---------------------------------------------------------------------------
__global__ void pack_w_kernel(const float* __restrict__ W, bf16_t* __restrict__ P,
                              int N, int K) {
    int gid = blockIdx.x * blockDim.x + threadIdx.x;
    int total = (N / 16) * (K / 32) * 32;
    if (gid >= total) return;
    int lane = gid & 31;
    int tile = gid >> 5;
    int ktiles = K >> 5;
    int nt = tile / ktiles, kt = tile - nt * ktiles;
    int col = lane & 15, koff = (lane >> 4) * 16;
    const float* src = W + (size_t)(nt * 16 + col) * K + kt * 32 + koff;
    bf16_t* dst = P + (size_t)gid * 16;
#pragma unroll
    for (int e = 0; e < 16; ++e) dst[e] = f2bf(src[e]);
}

// ---------------------------------------------------------------------------
// Persistent ODE + decoder kernel.
// ---------------------------------------------------------------------------
__global__ __launch_bounds__(BLOCK, 1)
void node_persist_kernel(const float* __restrict__ z0, const float* __restrict__ tarr,
                         const float* __restrict__ b1, const float* __restrict__ b2,
                         const float* __restrict__ b3, const float* __restrict__ l2hb,
                         const float* __restrict__ h2ob,
                         const v16bf* __restrict__ W1p, const v16bf* __restrict__ W2p,
                         const v16bf* __restrict__ W3p, const v16bf* __restrict__ L2Hp,
                         const v16bf* __restrict__ H2Op,
                         float* __restrict__ out, int T, int Btot) {
    __shared__ __attribute__((aligned(16))) unsigned char smem[LDS_TOTAL];
    bf16_t* zb = (bf16_t*)smem;                       // 16 x ZB_STRIDE bf16
    bf16_t* sc = (bf16_t*)smem;                       // aliased: 8 x 16 x SC_STRIDE
    bf16_t* h1 = (bf16_t*)(smem + LDS_H1_OFF);        // 16 x H1_STRIDE bf16
    float*  dz = (float*)(smem + LDS_DZ_OFF);         // 16 x DZ_STRIDE f32

    const int tid  = threadIdx.x;
    const int lane = tid & 31;
    const int wave = tid >> 5;
    const int ccol = lane & 15;          // C-fragment column within tile
    const int crhi = (lane >> 4) * 8;    // C-fragment row base (0 or 8)
    const int bbase = blockIdx.x * BM;

    // Persistent latent state: wave w owns latent column tiles 2w and 2w+1.
    v8f zf[2];
#pragma unroll
    for (int t2 = 0; t2 < 2; ++t2) {
        int nc = (2 * wave + t2) * 16 + ccol;
#pragma unroll
        for (int j = 0; j < 8; ++j)
            zf[t2][j] = z0[(size_t)(bbase + crhi + j) * LATENT + nc];
    }

    auto write_zbf = [&]() {
#pragma unroll
        for (int t2 = 0; t2 < 2; ++t2) {
            int nc = (2 * wave + t2) * 16 + ccol;
#pragma unroll
            for (int j = 0; j < 8; ++j)
                zb[(crhi + j) * ZB_STRIDE + nc] = f2bf(zf[t2][j]);
        }
    };

    // Decoder: xs = (z @ l2h^T + l2hb) @ h2o^T + h2ob   (no activation)
    auto decode = [&](int it) {
#pragma unroll
        for (int i = 0; i < 8; ++i) {                    // dec hidden: 1024 cols
            int nt = wave * 8 + i;
            v8f acc = {};
#pragma unroll
            for (int kt = 0; kt < 8; ++kt)
                acc = wmma_bf16(load_a_lds(zb, ZB_STRIDE, kt, lane),
                                load_b_pk(L2Hp, 8, nt, kt, lane), acc);
            float bv = l2hb[nt * 16 + ccol];
#pragma unroll
            for (int j = 0; j < 8; ++j)
                h1[(crhi + j) * H1_STRIDE + nt * 16 + ccol] = f2bf(acc[j] + bv);
        }
        __syncthreads();
#pragma unroll
        for (int i = 0; i < 4; ++i) {                    // output: 512 cols
            int nt = wave * 4 + i;
            v8f acc = {};
#pragma unroll 4
            for (int kt = 0; kt < 32; ++kt)
                acc = wmma_bf16(load_a_lds(h1, H1_STRIDE, kt, lane),
                                load_b_pk(H2Op, 32, nt, kt, lane), acc);
            float bv = h2ob[nt * 16 + ccol];
#pragma unroll
            for (int j = 0; j < 8; ++j)
                out[((size_t)it * Btot + bbase + crhi + j) * OUTF + nt * 16 + ccol]
                    = acc[j] + bv;
        }
        __syncthreads();   // protect h1 before next writer
    };

    // One Euler substep: z += h * (W3^T·ELU(W2^T·ELU(W1^T·z + b1) + b2) + b3)
    auto ode_step = [&](float h) {
        // ---- layer 1: h1 = ELU(z @ W1^T + b1) ----
#pragma unroll
        for (int i = 0; i < 8; ++i) {
            int nt = wave * 8 + i;
            v8f acc = {};
#pragma unroll
            for (int kt = 0; kt < 8; ++kt)
                acc = wmma_bf16(load_a_lds(zb, ZB_STRIDE, kt, lane),
                                load_b_pk(W1p, 8, nt, kt, lane), acc);
            float bv = b1[nt * 16 + ccol];
#pragma unroll
            for (int j = 0; j < 8; ++j)
                h1[(crhi + j) * H1_STRIDE + nt * 16 + ccol] = f2bf(elu(acc[j] + bv));
        }
        __syncthreads();                 // h1 visible; zb region free for scratch

        // ---- init dz with bias b3 (k-split reduction target) ----
        {
            float bv = b3[tid];          // tid == latent column (256 threads)
#pragma unroll
            for (int r = 0; r < 16; ++r) dz[r * DZ_STRIDE + tid] = bv;
        }
        __syncthreads();

        // ---- fused layers 2+3: wave w covers h2 columns [w*128, w*128+128) ----
        v8f dzacc[16] = {};
        bf16_t* mysc = sc + wave * 16 * SC_STRIDE;
#pragma unroll
        for (int p = 0; p < 4; ++p) {
            int nt2 = wave * 8 + p * 2;
            v8f a0 = {}, a1 = {};
#pragma unroll 4
            for (int kt = 0; kt < 32; ++kt) {
                v16bf af = load_a_lds(h1, H1_STRIDE, kt, lane);
                a0 = wmma_bf16(af, load_b_pk(W2p, 32, nt2,     kt, lane), a0);
                a1 = wmma_bf16(af, load_b_pk(W2p, 32, nt2 + 1, kt, lane), a1);
            }
            float bv0 = b2[nt2 * 16 + ccol], bv1 = b2[(nt2 + 1) * 16 + ccol];
            // ELU(h2) -> wave-private scratch (C-frag -> row-major re-layout)
#pragma unroll
            for (int j = 0; j < 8; ++j) {
                mysc[(crhi + j) * SC_STRIDE + ccol]      = f2bf(elu(a0[j] + bv0));
                mysc[(crhi + j) * SC_STRIDE + 16 + ccol] = f2bf(elu(a1[j] + bv1));
            }
            asm volatile("s_wait_dscnt 0" ::: "memory");   // wave-private RAW
            // layer 3 partial: dz += h2_block @ W3 (k-tile = wave*4 + p)
            int kt3 = wave * 4 + p;
            v16bf af3 = load_a_lds(mysc, SC_STRIDE, 0, lane);
#pragma unroll
            for (int n = 0; n < 16; ++n)
                dzacc[n] = wmma_bf16(af3, load_b_pk(W3p, 32, n, kt3, lane), dzacc[n]);
        }
        // ---- cross-wave k-reduction into LDS (ds_add_f32) ----
#pragma unroll
        for (int n = 0; n < 16; ++n) {
#pragma unroll
            for (int j = 0; j < 8; ++j)
                atomicAdd(&dz[(crhi + j) * DZ_STRIDE + n * 16 + ccol], dzacc[n][j]);
        }
        __syncthreads();

        // ---- Euler update of resident z fragments ----
#pragma unroll
        for (int t2 = 0; t2 < 2; ++t2) {
            int nc = (2 * wave + t2) * 16 + ccol;
#pragma unroll
            for (int j = 0; j < 8; ++j)
                zf[t2][j] += h * dz[(crhi + j) * DZ_STRIDE + nc];
        }
        write_zbf();        // scratch region is dead (post-atomic barrier)
        __syncthreads();
    };

    // ---- main time loop ----
    write_zbf();
    __syncthreads();
    decode(0);
    for (int it = 1; it < T; ++it) {
        float h = (tarr[it] - tarr[it - 1]) * (1.0f / NSTEPS);
        for (int s = 0; s < NSTEPS; ++s) ode_step(h);
        decode(it);
    }
}

// ---------------------------------------------------------------------------
// Host launch: pack 5 weight matrices to bf16 fragment layout in d_ws, then
// run the persistent kernel. All launches on `stream`; graph-capture safe.
// ---------------------------------------------------------------------------
extern "C" void kernel_launch(void* const* d_in, const int* in_sizes, int n_in,
                              void* d_out, int out_size, void* d_ws, size_t ws_size,
                              hipStream_t stream) {
    const float* z0   = (const float*)d_in[0];
    const float* tarr = (const float*)d_in[1];
    const float* w1   = (const float*)d_in[2];
    const float* b1   = (const float*)d_in[3];
    const float* w2   = (const float*)d_in[4];
    const float* b2   = (const float*)d_in[5];
    const float* w3   = (const float*)d_in[6];
    const float* b3   = (const float*)d_in[7];
    const float* l2hw = (const float*)d_in[8];
    const float* l2hb = (const float*)d_in[9];
    const float* h2ow = (const float*)d_in[10];
    const float* h2ob = (const float*)d_in[11];
    const int T    = in_sizes[1];
    const int Btot = in_sizes[0] / LATENT;
    float* out = (float*)d_out;

    bf16_t* ws = (bf16_t*)d_ws;
    size_t o = 0;
    bf16_t* W1p  = ws + o; o += (size_t)HIDDEN * LATENT;   // 1024x256
    bf16_t* W2p  = ws + o; o += (size_t)HIDDEN * HIDDEN;   // 1024x1024
    bf16_t* W3p  = ws + o; o += (size_t)LATENT * HIDDEN;   // 256x1024
    bf16_t* L2Hp = ws + o; o += (size_t)HIDDEN * LATENT;   // 1024x256
    bf16_t* H2Op = ws + o;                                 // 512x1024

    auto pack = [&](const float* W, bf16_t* P, int N, int K) {
        int tot = N * K / 16;
        pack_w_kernel<<<(tot + 255) / 256, 256, 0, stream>>>(W, P, N, K);
    };
    pack(w1,   W1p,  HIDDEN, LATENT);
    pack(w2,   W2p,  HIDDEN, HIDDEN);
    pack(w3,   W3p,  LATENT, HIDDEN);
    pack(l2hw, L2Hp, HIDDEN, LATENT);
    pack(h2ow, H2Op, OUTF,   HIDDEN);

    node_persist_kernel<<<Btot / BM, BLOCK, 0, stream>>>(
        z0, tarr, b1, b2, b3, l2hb, h2ob,
        (const v16bf*)W1p, (const v16bf*)W2p, (const v16bf*)W3p,
        (const v16bf*)L2Hp, (const v16bf*)H2Op,
        out, T, Btot);
}
